// GCNJK_47107201303139
// MI455X (gfx1250) — matrix operand order repaired and checked by reading, not compiled, not run on hardware
//
#include <hip/hip_runtime.h>

typedef __attribute__((ext_vector_type(16))) _Float16 v16h;
typedef __attribute__((ext_vector_type(8)))  float    v8f;

#define HDIM   128
#define KDIM   128
#define KCHUNK (KDIM / 32)   // 4 k-chunks of 32
#define BN_EPS 1e-5f

// ---------------------------------------------------------------- utilities
__global__ void zero_f32(float* __restrict__ p, int n) {
  int i = blockIdx.x * blockDim.x + threadIdx.x;
  if (i < n) p[i] = 0.0f;
}

__global__ void deg_accum(const int* __restrict__ dst, int E, float* __restrict__ deg) {
  int i = blockIdx.x * blockDim.x + threadIdx.x;
  if (i < E) atomicAdd(&deg[dst[i]], 1.0f);
}

__global__ void dinv_finalize(float* __restrict__ deg, int n) {
  int i = blockIdx.x * blockDim.x + threadIdx.x;
  if (i < n) deg[i] = rsqrtf(deg[i] + 1.0f);   // +1 = self loop; always > 0
}

// ------------------------------------------------- pack W into B-matrix swizzle
// Layout: for col-tile ct, k-chunk kc, lane l, half h (contiguous 16 halfs/lane):
//   B element (K = kc*32 + (h>=8 ? 16 : 0) + (l>=16 ? 8 : 0) + (h&7),  N = ct*16 + (l&15))
__global__ void pack_w(const float* __restrict__ W, _Float16* __restrict__ wp, int Hout) {
  int t = blockIdx.x * blockDim.x + threadIdx.x;
  int total = (Hout >> 4) * KCHUNK * 512;
  if (t >= total) return;
  int h    = t & 15;
  int lane = (t >> 4) & 31;
  int rest = t >> 9;
  int kc   = rest % KCHUNK;
  int ct   = rest / KCHUNK;
  int kg   = kc * 32 + ((h >= 8) ? 16 : 0) + ((lane >= 16) ? 8 : 0) + (h & 7);
  int col  = ct * 16 + (lane & 15);
  wp[t] = (_Float16)W[kg * Hout + col];
}

// ------------------------------------------------------------- WMMA GEMM
// Y[N x (NT*16)] = X[N x 128] * Wpack  (+ optional bias), f16 inputs, f32 acc.
// One wave per 16-row tile; NT 16-col tiles per wave.
template <int NT>
__global__ __launch_bounds__(256) void gemm_wmma(
    const float* __restrict__ X, const _Float16* __restrict__ Wp,
    float* __restrict__ Y, const float* __restrict__ bias, int nRows) {
  const int lane  = threadIdx.x & 31;
  const int wave  = threadIdx.x >> 5;
  const int tile  = blockIdx.x * (blockDim.x >> 5) + wave;
  const int nTile = (nRows + 15) >> 4;
  if (tile >= nTile) return;                 // wave-uniform: EXEC stays all-ones
  const int row0  = tile * 16;
  int rowA = row0 + (lane & 15);
  if (rowA >= nRows) rowA = nRows - 1;       // clamp (N=50000 is exact anyway)
  const int base8 = (lane >> 4) * 8;         // A-matrix K interleave per lane half

  v8f acc[NT];
#pragma unroll
  for (int j = 0; j < NT; ++j) acc[j] = (v8f)(0.0f);

  const float* xr = X + (size_t)rowA * KDIM;
#pragma unroll
  for (int kc = 0; kc < KCHUNK; ++kc) {
    const float* p = xr + kc * 32 + base8;
    const float4 p0 = *(const float4*)(p);
    const float4 p1 = *(const float4*)(p + 4);
    const float4 p2 = *(const float4*)(p + 16);
    const float4 p3 = *(const float4*)(p + 20);
    v16h a;
    a[0]  = (_Float16)p0.x; a[1]  = (_Float16)p0.y; a[2]  = (_Float16)p0.z; a[3]  = (_Float16)p0.w;
    a[4]  = (_Float16)p1.x; a[5]  = (_Float16)p1.y; a[6]  = (_Float16)p1.z; a[7]  = (_Float16)p1.w;
    a[8]  = (_Float16)p2.x; a[9]  = (_Float16)p2.y; a[10] = (_Float16)p2.z; a[11] = (_Float16)p2.w;
    a[12] = (_Float16)p3.x; a[13] = (_Float16)p3.y; a[14] = (_Float16)p3.z; a[15] = (_Float16)p3.w;
#pragma unroll
    for (int j = 0; j < NT; ++j) {
      const v16h b = *(const v16h*)(Wp + (((j * KCHUNK + kc) * 32 + lane) << 4));
      acc[j] = __builtin_amdgcn_wmma_f32_16x16x32_f16(
          false, a, false, b, (short)0, acc[j], false, false);
    }
  }

  // C/D layout: lane 0-15 -> M = v, lane 16-31 -> M = 8+v, N = lane&15
  const int Hout   = NT * 16;
  const int colL   = lane & 15;
  const int rowAdd = (lane >> 4) * 8;
#pragma unroll
  for (int j = 0; j < NT; ++j) {
    const int col = j * 16 + colL;
    const float bv = bias ? bias[col] : 0.0f;
#pragma unroll
    for (int v = 0; v < 8; ++v) {
      const int r = row0 + rowAdd + v;
      if (r < nRows) Y[(size_t)r * Hout + col] = acc[j][v] + bv;
    }
  }
}

// ------------------------------------------- agg = bias + hW * dinv^2 (self loop)
__global__ void agg_init(const float* __restrict__ hW, const float* __restrict__ dinv,
                         const float* __restrict__ bias, float* __restrict__ agg, int n) {
  int t = blockIdx.x * blockDim.x + threadIdx.x;
  if (t >= n * (HDIM / 4)) return;
  int i  = t >> 5;                 // HDIM/4 == 32
  int c4 = (t & 31) * 4;
  float di = dinv[i];
  float s  = di * di;
  const float4 hv = *(const float4*)(hW + (size_t)i * HDIM + c4);
  const float4 bv = *(const float4*)(bias + c4);
  float4 o;
  o.x = bv.x + hv.x * s; o.y = bv.y + hv.y * s;
  o.z = bv.z + hv.z * s; o.w = bv.w + hv.w * s;
  *(float4*)(agg + (size_t)i * HDIM + c4) = o;
}

// ---------------------------------------- one wave per edge, float4 atomics/lane
__global__ void scatter_edges(const float* __restrict__ hW, const int* __restrict__ src,
                              const int* __restrict__ dst, const float* __restrict__ dinv,
                              float* __restrict__ agg, int E) {
  int g = blockIdx.x * blockDim.x + threadIdx.x;
  int e = g >> 5;
  if (e >= E) return;
  int c = (g & 31) * 4;
  int s = src[e], d = dst[e];
  float nw = dinv[s] * dinv[d];
  const float4 hv = *(const float4*)(hW + (size_t)s * HDIM + c);
  float* ap = agg + (size_t)d * HDIM + c;
  atomicAdd(ap + 0, hv.x * nw);
  atomicAdd(ap + 1, hv.y * nw);
  atomicAdd(ap + 2, hv.z * nw);
  atomicAdd(ap + 3, hv.w * nw);
}

// ------------------------------------------------------------- BatchNorm
__global__ void bn_stats(const float* __restrict__ agg, float* __restrict__ sums, int n) {
  __shared__ float s1[256], s2[256];
  int c    = threadIdx.x & 127;
  int half = threadIdx.x >> 7;
  int rpb  = (n + gridDim.x - 1) / gridDim.x;
  int r0   = blockIdx.x * rpb;
  int r1   = r0 + rpb; if (r1 > n) r1 = n;
  float a = 0.0f, b = 0.0f;
  for (int i = r0 + half; i < r1; i += 2) {
    float v = agg[(size_t)i * HDIM + c];
    a += v; b += v * v;
  }
  s1[threadIdx.x] = a; s2[threadIdx.x] = b;
  __syncthreads();
  if (half == 0) {
    atomicAdd(&sums[c],        a + s1[c + 128]);
    atomicAdd(&sums[128 + c],  b + s2[c + 128]);
  }
}

__global__ void bn_finalize(const float* __restrict__ sums, const float* __restrict__ gamma,
                            const float* __restrict__ beta, float* __restrict__ coef, int n) {
  int c = threadIdx.x;
  if (c >= HDIM) return;
  float invN = 1.0f / (float)n;
  float m = sums[c] * invN;
  float v = sums[HDIM + c] * invN - m * m;          // biased var
  float a = gamma[c] * rsqrtf(v + BN_EPS);
  coef[c]        = a;
  coef[HDIM + c] = beta[c] - m * a;
}

__global__ void bn_apply_relu(const float* __restrict__ agg, const float* __restrict__ coef,
                              float* __restrict__ out, int n) {
  int t = blockIdx.x * blockDim.x + threadIdx.x;
  if (t >= n * (HDIM / 4)) return;
  int i  = t >> 5;
  int c4 = (t & 31) * 4;
  const float4 v = *(const float4*)(agg + (size_t)i * HDIM + c4);
  const float4 a = *(const float4*)(coef + c4);
  const float4 b = *(const float4*)(coef + HDIM + c4);
  float4 o;
  o.x = fmaxf(0.0f, a.x * v.x + b.x);
  o.y = fmaxf(0.0f, a.y * v.y + b.y);
  o.z = fmaxf(0.0f, a.z * v.z + b.z);
  o.w = fmaxf(0.0f, a.w * v.w + b.w);
  *(float4*)(out + (size_t)i * HDIM + c4) = o;
}

__global__ void jk_max3(const float* __restrict__ a, const float* __restrict__ b,
                        const float* __restrict__ c, float* __restrict__ o, int n4) {
  int t = blockIdx.x * blockDim.x + threadIdx.x;
  if (t >= n4) return;
  const float4 va = ((const float4*)a)[t];
  const float4 vb = ((const float4*)b)[t];
  const float4 vc = ((const float4*)c)[t];
  float4 r;
  r.x = fmaxf(fmaxf(va.x, vb.x), vc.x);
  r.y = fmaxf(fmaxf(va.y, vb.y), vc.y);
  r.z = fmaxf(fmaxf(va.z, vb.z), vc.z);
  r.w = fmaxf(fmaxf(va.w, vb.w), vc.w);
  ((float4*)o)[t] = r;
}

// ---------------------------------------------------------------- launch
static inline size_t alignup(size_t x) { return (x + 255) & ~(size_t)255; }

extern "C" void kernel_launch(void* const* d_in, const int* in_sizes, int n_in,
                              void* d_out, int out_size, void* d_ws, size_t ws_size,
                              hipStream_t stream) {
  const float* x   = (const float*)d_in[0];
  const int*   ei  = (const int*)d_in[1];
  const float* W1  = (const float*)d_in[2];
  const float* b1  = (const float*)d_in[3];
  const float* g1  = (const float*)d_in[4];
  const float* be1 = (const float*)d_in[5];
  const float* W2  = (const float*)d_in[6];
  const float* b2  = (const float*)d_in[7];
  const float* g2  = (const float*)d_in[8];
  const float* be2 = (const float*)d_in[9];
  const float* W3  = (const float*)d_in[10];
  const float* b3  = (const float*)d_in[11];
  const float* Wp  = (const float*)d_in[12];
  const float* bp  = (const float*)d_in[13];

  const int N = in_sizes[0] / HDIM;
  const int E = in_sizes[1] / 2;
  const int OUT = in_sizes[12] / HDIM;        // 64
  const int* src = ei;
  const int* dst = ei + E;

  // workspace carve-up
  char* w = (char*)d_ws;
  float*    dinv  = (float*)w;                 w += alignup((size_t)N * 4);
  float*    hW    = (float*)w;                 w += alignup((size_t)N * HDIM * 4);
  float*    agg   = (float*)w;                 w += alignup((size_t)N * HDIM * 4);
  float*    h1    = (float*)w;                 w += alignup((size_t)N * HDIM * 4);
  float*    h2    = (float*)w;                 w += alignup((size_t)N * HDIM * 4);
  _Float16* wpack = (_Float16*)w;              w += alignup((size_t)HDIM * HDIM * 2);
  float*    sums  = (float*)w;                 w += alignup(256 * 4);
  float*    coef  = (float*)w;                 w += alignup(256 * 4);
  (void)ws_size; (void)n_in; (void)out_size;

  const int nTiles    = (N + 15) / 16;
  const dim3 gGemm((nTiles + 7) / 8);          // 8 waves / block
  const int gElem  = (N * 32 + 255) / 256;     // N*HDIM/4 threads
  const int gScat  = (E * 32 + 255) / 256;     // 1 wave / edge
  const int gPackH = (8 * KCHUNK * 512 + 255) / 256;
  const int gPackO = ((OUT / 16) * KCHUNK * 512 + 255) / 256;

  // degrees -> dinv
  zero_f32<<<(N + 255) / 256, 256, 0, stream>>>(dinv, N);
  deg_accum<<<(E + 255) / 256, 256, 0, stream>>>(dst, E, dinv);
  dinv_finalize<<<(N + 255) / 256, 256, 0, stream>>>(dinv, N);

  // ---- layer 1: conv -> BN -> ReLU
  pack_w<<<gPackH, 256, 0, stream>>>(W1, wpack, HDIM);
  gemm_wmma<8><<<gGemm, 256, 0, stream>>>(x, wpack, hW, nullptr, N);
  agg_init<<<gElem, 256, 0, stream>>>(hW, dinv, b1, agg, N);
  scatter_edges<<<gScat, 256, 0, stream>>>(hW, src, dst, dinv, agg, E);
  zero_f32<<<1, 256, 0, stream>>>(sums, 256);
  bn_stats<<<256, 256, 0, stream>>>(agg, sums, N);
  bn_finalize<<<1, 128, 0, stream>>>(sums, g1, be1, coef, N);
  bn_apply_relu<<<gElem, 256, 0, stream>>>(agg, coef, h1, N);

  // ---- layer 2: conv -> BN -> ReLU
  pack_w<<<gPackH, 256, 0, stream>>>(W2, wpack, HDIM);
  gemm_wmma<8><<<gGemm, 256, 0, stream>>>(h1, wpack, hW, nullptr, N);
  agg_init<<<gElem, 256, 0, stream>>>(hW, dinv, b2, agg, N);
  scatter_edges<<<gScat, 256, 0, stream>>>(hW, src, dst, dinv, agg, E);
  zero_f32<<<1, 256, 0, stream>>>(sums, 256);
  bn_stats<<<256, 256, 0, stream>>>(agg, sums, N);
  bn_finalize<<<1, 128, 0, stream>>>(sums, g2, be2, coef, N);
  bn_apply_relu<<<gElem, 256, 0, stream>>>(agg, coef, h2, N);

  // ---- layer 3: conv only (h3 lives in agg)
  pack_w<<<gPackH, 256, 0, stream>>>(W3, wpack, HDIM);
  gemm_wmma<8><<<gGemm, 256, 0, stream>>>(h2, wpack, hW, nullptr, N);
  // careful: agg_init reads hW, scatter reads hW; re-init agg after gemm
  agg_init<<<gElem, 256, 0, stream>>>(hW, dinv, b3, agg, N);
  scatter_edges<<<gScat, 256, 0, stream>>>(hW, src, dst, dinv, agg, E);

  // ---- JumpingKnowledge max -> reuse hW as jk buffer
  jk_max3<<<gElem, 256, 0, stream>>>(h1, h2, agg, hW, N * 32);

  // ---- final projection: d_out = jk @ Wp + bp   (N x 64, fp32)
  pack_w<<<gPackO, 256, 0, stream>>>(Wp, wpack, OUT);
  gemm_wmma<4><<<gGemm, 256, 0, stream>>>(hW, wpack, (float*)d_out, bp, N);
}
